// RelationalGraphletConvolutionGroupAttn_63977832841260
// MI455X (gfx1250) — matrix operand order repaired and checked by reading, not compiled, not run on hardware
//
#include <hip/hip_runtime.h>

// ---------------------------------------------------------------------------
// RelationalGraphletConvolutionGroupAttn — CDNA5 / gfx1250 implementation
// B=32, N=4096, D=256, KD=128, Q=256(G=64,g=4), R=16, P=16 (RP=256), F=64
// Heavy GEMMs: v_wmma_f32_16x16x32_bf16, 16x64 strip per wave (4x B-blocking).
// ---------------------------------------------------------------------------

#define BB   32
#define NN   4096
#define DD   256
#define KDIM 128
#define QQ   256
#define GG   64
#define GLET 4
#define RP   256     // R*P
#define FF   64

typedef __attribute__((ext_vector_type(16))) __bf16 v16bf;
typedef __attribute__((ext_vector_type(8)))  float  v8f;

union FragU { v16bf v; uint4 u4[2]; };

__device__ __forceinline__ unsigned short f32_to_bf16(float f) {
  unsigned int u = __float_as_uint(f);
  u += 0x7fffu + ((u >> 16) & 1u);   // round-to-nearest-even
  return (unsigned short)(u >> 16);
}

// A fragment: 16x32 bf16, A row-major (lda in elements).
// lanes 0-15: row M=lane, K {k0..k0+7} in V0-3, {k0+16..k0+23} in V4-7;
// lanes 16-31: same rows, K {+8..+15}, {+24..+31}.  -> two 16B contiguous loads.
__device__ __forceinline__ v16bf load_a_frag(const unsigned short* A, int lda,
                                             int m0, int k0, int lane) {
  int r = lane & 15, h = lane >> 4;
  const unsigned short* p = A + (size_t)(m0 + r) * lda + k0 + h * 8;
  FragU f;
  f.u4[0] = *(const uint4*)(p);
  f.u4[1] = *(const uint4*)(p + 16);
  return f.v;
}

// B fragment: 32x16 bf16 supplied as Bt = B^T row-major (N x K, ldb in elems).
// lanes 0-15: col N=lane, K {k0..k0+7},{k0+16..k0+23}; lanes 16-31: K {+8..+15},{+24..+31}.
__device__ __forceinline__ v16bf load_b_frag(const unsigned short* Bt, int ldb,
                                             int n0, int k0, int lane) {
  int r = lane & 15, h = lane >> 4;
  const unsigned short* p = Bt + (size_t)(n0 + r) * ldb + k0 + h * 8;
  FragU f;
  f.u4[0] = *(const uint4*)(p);
  f.u4[1] = *(const uint4*)(p + 16);
  return f.v;
}

// ----------------------------- prep kernels --------------------------------

// Qp[q,d] = sum_c Gq[q,c] * Wk[d,c]   -> bf16 (256x256)
__global__ void k_prep_qp(const float* __restrict__ gq, const float* __restrict__ wk,
                          unsigned short* __restrict__ qp) {
  int idx = blockIdx.x * blockDim.x + threadIdx.x;   // 65536
  int q = idx >> 8, d = idx & 255;
  const float* gr = gq + q * KDIM;
  const float* wr = wk + d * KDIM;
  float s = 0.f;
#pragma unroll 8
  for (int c = 0; c < KDIM; ++c) s += gr[c] * wr[c];
  qp[idx] = f32_to_bf16(s);
}

// cpos[q,n] = sum_c Gq[q,c] * pos[n,c]   -> f32 (256x4096)
__global__ void k_prep_cpos(const float* __restrict__ gq, const float* __restrict__ pos,
                            float* __restrict__ cpos) {
  int idx = blockIdx.x * blockDim.x + threadIdx.x;   // 1048576
  int q = idx >> 12, n = idx & 4095;
  const float* gr = gq + q * KDIM;
  const float* pr = pos + n * KDIM;
  float s = 0.f;
#pragma unroll 8
  for (int c = 0; c < KDIM; ++c) s += gr[c] * pr[c];
  cpos[(size_t)q * NN + n] = s;
}

// W2t[rp, d] = bf16(W_mdipr[d, rp])  (transpose for Bt convention)
__global__ void k_prep_w2t(const float* __restrict__ wm, unsigned short* __restrict__ w2t) {
  int idx = blockIdx.x * blockDim.x + threadIdx.x;   // 65536
  int rp = idx >> 8, d = idx & 255;
  w2t[(size_t)rp * DD + d] = f32_to_bf16(wm[(size_t)d * RP + rp]);
}

// xb[b,n,d] = bf16(x);  xT[b,d,n] = bf16(x)
__global__ void k_pack_x(const float* __restrict__ x, unsigned short* __restrict__ xb,
                         unsigned short* __restrict__ xT) {
  size_t idx = (size_t)blockIdx.x * blockDim.x + threadIdx.x;  // B*N*D
  int d = (int)(idx & (DD - 1));
  size_t bn = idx >> 8;               // b*N + n
  int n = (int)(bn & (NN - 1));
  int b = (int)(bn >> 12);
  unsigned short v = f32_to_bf16(x[idx]);
  xb[idx] = v;
  xT[((size_t)b * DD + d) * NN + n] = v;
}

// ----------------------------- WMMA GEMMs ----------------------------------
// Each wave computes a 16x64 strip: 1 A-fragment reused across 4 B-fragments
// per K step (4x less A traffic, 4 v_wmma per A load).

// L[b,q,n] = beta * (sum_d Qp[q,d] * xb[b,n,d] + cpos[q,n])
__global__ void k_logits(const unsigned short* __restrict__ qp,
                         const unsigned short* __restrict__ xb,
                         const float* __restrict__ cpos,
                         float* __restrict__ L, float beta) {
  const int NG = NN / 64, MT = QQ / 16;            // 64 strip-groups, 16 m-tiles
  int lane = threadIdx.x & 31;
  int strip = blockIdx.x * (blockDim.x >> 5) + (threadIdx.x >> 5);
  int b  = strip / (MT * NG);
  int t  = strip % (MT * NG);
  int m0 = (t / NG) * 16, n0 = (t % NG) * 64;
  const unsigned short* Bt = xb + (size_t)b * NN * DD;
  v8f acc[4];
#pragma unroll
  for (int j = 0; j < 4; ++j) acc[j] = (v8f){0.f,0.f,0.f,0.f,0.f,0.f,0.f,0.f};
#pragma unroll
  for (int k0 = 0; k0 < DD; k0 += 32) {
    v16bf a = load_a_frag(qp, DD, m0, k0, lane);
#pragma unroll
    for (int j = 0; j < 4; ++j) {
      v16bf bf = load_b_frag(Bt, DD, n0 + 16 * j, k0, lane);
      acc[j] = __builtin_amdgcn_wmma_f32_16x16x32_bf16(false, a, false, bf,
                                                       (short)0, acc[j], false, false);
    }
  }
  int row0 = m0 + ((lane >> 4) << 3);
  float* Lb = L + (size_t)b * QQ * NN;
#pragma unroll
  for (int j = 0; j < 4; ++j) {
    int col = n0 + 16 * j + (lane & 15);
#pragma unroll
    for (int v = 0; v < 8; ++v) {
      int row = row0 + v;
      Lb[(size_t)row * NN + col] = beta * (acc[j][v] + cpos[(size_t)row * NN + col]);
    }
  }
}

// row softmax over N=4096; rewrite probabilities in place as bf16
// (row-private: bf16 row occupies the first half of its own f32 row storage)
__global__ void k_softmax(float* __restrict__ L) {
  int row = blockIdx.x;                  // b*Q + q, 8192 rows
  float* Lr = L + (size_t)row * NN;
  int t = threadIdx.x;                   // 256 threads x 16 elems
  float vals[16];
  float m = -3.402823466e38f;
#pragma unroll
  for (int i = 0; i < 16; ++i) { vals[i] = Lr[t * 16 + i]; m = fmaxf(m, vals[i]); }
  __shared__ float red[256];
  red[t] = m; __syncthreads();
  for (int s = 128; s > 0; s >>= 1) { if (t < s) red[t] = fmaxf(red[t], red[t + s]); __syncthreads(); }
  float rowmax = red[0]; __syncthreads();
  float sum = 0.f;
#pragma unroll
  for (int i = 0; i < 16; ++i) { vals[i] = __expf(vals[i] - rowmax); sum += vals[i]; }
  red[t] = sum; __syncthreads();
  for (int s = 128; s > 0; s >>= 1) { if (t < s) red[t] += red[t + s]; __syncthreads(); }
  float inv = 1.f / red[0];
  unsigned short* Pr = (unsigned short*)Lr;
#pragma unroll
  for (int i = 0; i < 16; ++i) Pr[t * 16 + i] = f32_to_bf16(vals[i] * inv);
}

// z[b,q,d] = sum_n P[b,q,n] * x[b,n,d]  -> bf16   (A = P bf16 lda=8192, Bt = xT)
__global__ void k_attn(const unsigned short* __restrict__ Pbf,
                       const unsigned short* __restrict__ xT,
                       unsigned short* __restrict__ zb) {
  const int NG = DD / 64, MT = QQ / 16;            // 4, 16
  int lane = threadIdx.x & 31;
  int strip = blockIdx.x * (blockDim.x >> 5) + (threadIdx.x >> 5);
  int b  = strip / (MT * NG);
  int t  = strip % (MT * NG);
  int m0 = (t / NG) * 16, n0 = (t % NG) * 64;
  const unsigned short* A  = Pbf + (size_t)b * QQ * (2 * NN);   // row stride 8192 ushorts
  const unsigned short* Bt = xT + (size_t)b * DD * NN;
  v8f acc[4];
#pragma unroll
  for (int j = 0; j < 4; ++j) acc[j] = (v8f){0.f,0.f,0.f,0.f,0.f,0.f,0.f,0.f};
#pragma unroll 2
  for (int k0 = 0; k0 < NN; k0 += 32) {
    v16bf a = load_a_frag(A, 2 * NN, m0, k0, lane);
#pragma unroll
    for (int j = 0; j < 4; ++j) {
      v16bf bf = load_b_frag(Bt, NN, n0 + 16 * j, k0, lane);
      acc[j] = __builtin_amdgcn_wmma_f32_16x16x32_bf16(false, a, false, bf,
                                                       (short)0, acc[j], false, false);
    }
  }
  int row0 = m0 + ((lane >> 4) << 3);
  unsigned short* Zb = zb + (size_t)b * QQ * DD;
#pragma unroll
  for (int j = 0; j < 4; ++j) {
    int col = n0 + 16 * j + (lane & 15);
#pragma unroll
    for (int v = 0; v < 8; ++v)
      Zb[(size_t)(row0 + v) * DD + col] = f32_to_bf16(acc[j][v]);
  }
}

// p[m, rp] = sum_d z[m,d] * W_mdipr[d, rp]   (M = B*G*g = 8192) -> f32
__global__ void k_mdipr(const unsigned short* __restrict__ zb,
                        const unsigned short* __restrict__ w2t,
                        float* __restrict__ pbuf) {
  const int NG = RP / 64;                          // 4
  int lane = threadIdx.x & 31;
  int strip = blockIdx.x * (blockDim.x >> 5) + (threadIdx.x >> 5);
  int m0 = (strip / NG) * 16, n0 = (strip % NG) * 64;
  v8f acc[4];
#pragma unroll
  for (int j = 0; j < 4; ++j) acc[j] = (v8f){0.f,0.f,0.f,0.f,0.f,0.f,0.f,0.f};
#pragma unroll
  for (int k0 = 0; k0 < DD; k0 += 32) {
    v16bf a = load_a_frag(zb, DD, m0, k0, lane);
#pragma unroll
    for (int j = 0; j < 4; ++j) {
      v16bf bf = load_b_frag(w2t, DD, n0 + 16 * j, k0, lane);
      acc[j] = __builtin_amdgcn_wmma_f32_16x16x32_bf16(false, a, false, bf,
                                                       (short)0, acc[j], false, false);
    }
  }
  int row0 = m0 + ((lane >> 4) << 3);
#pragma unroll
  for (int j = 0; j < 4; ++j) {
    int col = n0 + 16 * j + (lane & 15);
#pragma unroll
    for (int v = 0; v < 8; ++v)
      pbuf[(size_t)(row0 + v) * RP + col] = acc[j][v];
  }
}

// Rt[i,j,r] = sum_p p[i,r,p]*p[j,r,p];  out[bg,f] = sum_{ijr} Rt * filters[f,ijr]
__global__ void k_finalize(const float* __restrict__ pbuf,
                           const float* __restrict__ filt,
                           float* __restrict__ out) {
  int bg = blockIdx.x;                   // 2048 blocks
  __shared__ float pl[GLET * RP];        // 4 x 256
  __shared__ float Rt[GLET * GLET * 16]; // 256
  int t = threadIdx.x;                   // 256
  const float* pb = pbuf + (size_t)bg * GLET * RP;
#pragma unroll
  for (int i = t; i < GLET * RP; i += 256) pl[i] = pb[i];
  __syncthreads();
  int i = t >> 6, j = (t >> 4) & 3, r = t & 15;
  float s = 0.f;
#pragma unroll
  for (int pp = 0; pp < 16; ++pp)
    s += pl[i * RP + r * 16 + pp] * pl[j * RP + r * 16 + pp];
  Rt[t] = s;                             // t == i*64 + j*16 + r
  __syncthreads();
  if (t < FF) {
    const float* fr = filt + t * 256;
    float a = 0.f;
#pragma unroll 8
    for (int e = 0; e < 256; ++e) a += Rt[e] * fr[e];
    out[(size_t)bg * FF + t] = a;
  }
}

// ----------------------------- launcher ------------------------------------

extern "C" void kernel_launch(void* const* d_in, const int* in_sizes, int n_in,
                              void* d_out, int out_size, void* d_ws, size_t ws_size,
                              hipStream_t stream) {
  const float* x    = (const float*)d_in[0];   // (B,N,D)
  const float* gq   = (const float*)d_in[1];   // (Q,KD)
  const float* pos  = (const float*)d_in[2];   // (N,KD)
  const float* wk   = (const float*)d_in[3];   // (D,KD)
  const float* wm   = (const float*)d_in[4];   // (D,R,P)
  const float* filt = (const float*)d_in[5];   // (F,g,g,R)
  float* out = (float*)d_out;                  // (B,G,F)
  char* ws = (char*)d_ws;

  const size_t sz_xb   = (size_t)BB * NN * DD * 2;      // 64 MiB
  const size_t sz_xT   = sz_xb;                         // 64 MiB
  const size_t sz_L    = (size_t)BB * QQ * NN * 4;      // 128 MiB (logits, then bf16 probs)
  const size_t sz_qp   = (size_t)QQ * DD * 2;
  const size_t sz_cpos = (size_t)QQ * NN * 4;
  const size_t sz_zb   = (size_t)BB * QQ * DD * 2;
  const size_t sz_w2t  = (size_t)RP * DD * 2;

  size_t off = 0;
  unsigned short* xb   = (unsigned short*)(ws + off); off += sz_xb;
  unsigned short* xT   = (unsigned short*)(ws + off); off += sz_xT;
  float*          L    = (float*)         (ws + off); off += sz_L;
  unsigned short* qp   = (unsigned short*)(ws + off); off += sz_qp;
  float*          cpos = (float*)         (ws + off); off += sz_cpos;
  unsigned short* zb   = (unsigned short*)(ws + off); off += sz_zb;
  unsigned short* w2t  = (unsigned short*)(ws + off); off += sz_w2t;
  float*          pbuf = (float*)         (ws + off);

  const float beta = 0.08838834764831845f;   // 128^-0.5

  // prep
  k_prep_qp  <<<(QQ * DD) / 256, 256, 0, stream>>>(gq, wk, qp);
  k_prep_cpos<<<(QQ * NN) / 256, 256, 0, stream>>>(gq, pos, cpos);
  k_prep_w2t <<<(RP * DD) / 256, 256, 0, stream>>>(wm, w2t);
  k_pack_x   <<<(BB * NN * DD) / 256, 256, 0, stream>>>(x, xb, xT);

  // fused key-proj + score GEMM: 32*16*64 = 32768 strips, 8 waves/block
  k_logits<<<(BB * (QQ / 16) * (NN / 64)) / 8, 256, 0, stream>>>(qp, xb, cpos, L, beta);

  // softmax rows, probabilities rewritten in place as bf16
  k_softmax<<<BB * QQ, 256, 0, stream>>>(L);

  // attention GEMM (K = 4096): 32*16*4 = 2048 strips
  k_attn<<<(BB * (QQ / 16) * (DD / 64)) / 8, 256, 0, stream>>>((const unsigned short*)L, xT, zb);

  // MDIPR projection GEMM: 512*4 = 2048 strips
  k_mdipr<<<((BB * QQ / 16) * (RP / 64)) / 8, 256, 0, stream>>>(zb, w2t, pbuf);

  // Gram matrices + filter contraction
  k_finalize<<<BB * GG, 256, 0, stream>>>(pbuf, filt, out);
}
